// MultiHeadAttention_25692494365207
// MI455X (gfx1250) — compile-verified, matrix operand order checked by reading, not run on hardware
//
#include <hip/hip_runtime.h>
#include <stdint.h>

// ---------------- problem constants ----------------
#define BB 2
#define NN 2048
#define EE 1024
#define HH 16
#define DD 64
#define MM (BB * NN)          // 4096 token rows

// ---------------- vector types ----------------
typedef __attribute__((ext_vector_type(16))) _Float16 v16h;
typedef __attribute__((ext_vector_type(8)))  _Float16 v8h;
typedef __attribute__((ext_vector_type(2)))  _Float16 v2h;
typedef __attribute__((ext_vector_type(8)))  float    v8f;
typedef __attribute__((ext_vector_type(4)))  float    v4f;
typedef __attribute__((ext_vector_type(4)))  uint32_t v4u;
typedef __attribute__((ext_vector_type(8)))  int      v8i;
typedef __attribute__((ext_vector_type(4)))  int      v4i;

// ---------------- WMMA helper ----------------
__device__ __forceinline__ v8f wmma_f16(v16h a, v16h b, v8f c) {
  // D = A(16x32) * B(32x16) + C(16x16), f32 accumulate
  return __builtin_amdgcn_wmma_f32_16x16x32_f16(false, a, false, b, (short)0, c,
                                                false, false);
}

// Load a 16x32 f16 A-fragment (or symmetric 32x16 B-fragment) from a
// row-major tile.  ISA 7.12.2: lanes 0-15 hold row (lane&15) with K-chunks
// {0..7,16..23}; lanes 16-31 hold the same rows with K-chunks {8..15,24..31}.
__device__ __forceinline__ v16h frag_ld(const _Float16* p0, int pitch) {
  const int lane = threadIdx.x & 31;
  const int sel  = lane >> 4;
  const _Float16* p = p0 + (size_t)(lane & 15) * pitch + sel * 8;
  union { v16h v; v8h h[2]; } u;
  u.h[0] = *(const v8h*)(p);
  u.h[1] = *(const v8h*)(p + 16);
  return u.v;
}

// ---------------- Tensor Data Mover (2D tile -> LDS) ----------------
__device__ __forceinline__ uint32_t lds_offset(const void* p) {
  // low 32 bits of a shared-aperture generic pointer = LDS byte offset
  return (uint32_t)(uintptr_t)p;
}

// 2D tile of 16-bit elements: tile_d0 contiguous elems per row, tile_d1 rows,
// row stride stride0 (elements).  pad_int/pad_amt are the raw D# codes
// (pad after 2^(1+pad_int) DWORDs, insert pad_amt+1 DWORDs); pad_en gates.
__device__ __forceinline__ void tdm_load_2d_f16(uint32_t lds_addr, const void* gptr,
                                                uint32_t tile_d0, uint32_t tile_d1,
                                                uint32_t tensor_d0, uint32_t tensor_d1,
                                                uint64_t stride0,
                                                uint32_t pad_int, uint32_t pad_amt,
                                                uint32_t pad_en) {
  const uint64_t ga = (uint64_t)(uintptr_t)gptr;
  v4u g0;
  g0.x = 1u;                                        // count = 1 valid descriptor
  g0.y = lds_addr;                                  // LDS byte address
  g0.z = (uint32_t)ga;                              // global_addr[31:0]
  g0.w = (uint32_t)((ga >> 32) & 0x01ffffffu) | (2u << 30);  // addr[56:32]|type=2

  // group1 bit layout (little endian over 4x u64):
  // [17:16]=data_size(1=2B) [20]=pad_en [24:22]=pad_interval [31:25]=pad_amount
  // [79:48]=tensor_dim0 [111:80]=tensor_dim1 [127:112]=tile_dim0
  // [143:128]=tile_dim1 [207:160]=tensor_dim0_stride
  uint64_t q0 = ((uint64_t)1u << 16)
              | ((uint64_t)(pad_en & 1u) << 20)
              | ((uint64_t)(pad_int & 7u) << 22)
              | ((uint64_t)(pad_amt & 0x7fu) << 25)
              | ((uint64_t)tensor_d0 << 48);
  uint64_t q1 = ((uint64_t)tensor_d0 >> 16)
              | ((uint64_t)tensor_d1 << 16)
              | ((uint64_t)(tile_d0 & 0xffffu) << 48);
  uint64_t q2 = (uint64_t)(tile_d1 & 0xffffu)
              | ((stride0 & 0xffffffffull) << 32);
  uint64_t q3 = (stride0 >> 32) & 0xffffull;

  v8i g1;
  g1[0] = (int)(uint32_t)q0; g1[1] = (int)(uint32_t)(q0 >> 32);
  g1[2] = (int)(uint32_t)q1; g1[3] = (int)(uint32_t)(q1 >> 32);
  g1[4] = (int)(uint32_t)q2; g1[5] = (int)(uint32_t)(q2 >> 32);
  g1[6] = (int)(uint32_t)q3; g1[7] = (int)(uint32_t)(q3 >> 32);
  v4i g2 = {0, 0, 0, 0};
  v4i g3 = {0, 0, 0, 0};
  v8i g4 = {0, 0, 0, 0, 0, 0, 0, 0};   // extra group (clang-23 6-arg form)
  __builtin_amdgcn_tensor_load_to_lds(g0, g1, g2, g3, g4, 0);
}

// =====================================================================
// Projection GEMM:  Y[M,E] = X[M,E] * W[E,E]^T + bias
// IN_F16:   X is f16 (attention output) vs f32
// TRANS_OUT: write f16 transposed per head: vT[b][col][token]
// OUT_F32:  write f32 (final projection into d_out), else f16 row-major
// Block: 256 thr (8 waves), tile 128x128, k-step 32, wave tile 32x64.
// =====================================================================
template <bool IN_F16, bool TRANS_OUT, bool OUT_F32>
__global__ __launch_bounds__(256) void proj_gemm_kernel(
    const void* __restrict__ xin, const float* __restrict__ W,
    const float* __restrict__ bias, void* __restrict__ out) {
  constexpr int KS = 32;
  constexpr int AP = 40;  // LDS pitch in halves (80B rows -> conflict friendly)
  __shared__ __attribute__((aligned(16))) _Float16 As[128 * AP];
  __shared__ __attribute__((aligned(16))) _Float16 Ws[128 * AP];

  const int m0 = blockIdx.x * 128;
  const int n0 = blockIdx.y * 128;
  const int t = threadIdx.x;
  const int lane = t & 31, w = t >> 5;
  const int wm = (w >> 1) * 32;   // wave row offset in block tile
  const int wn = (w & 1) * 64;    // wave col offset in block tile
  const int sel = lane >> 4, lr = lane & 15;

  v8f acc[2][4];
#pragma unroll
  for (int mi = 0; mi < 2; ++mi)
#pragma unroll
    for (int ni = 0; ni < 4; ++ni) acc[mi][ni] = (v8f){0, 0, 0, 0, 0, 0, 0, 0};

  const int srow = t >> 1;          // 0..127 staging row
  const int scol = (t & 1) * 16;    // 0 / 16

  for (int k0 = 0; k0 < EE; k0 += KS) {
    // ---- stage A tile (with fp32->f16 convert if needed) ----
    if constexpr (IN_F16) {
      const _Float16* src = (const _Float16*)xin + (size_t)(m0 + srow) * EE + k0 + scol;
      *(v8h*)&As[srow * AP + scol]     = *(const v8h*)(src);
      *(v8h*)&As[srow * AP + scol + 8] = *(const v8h*)(src + 8);
    } else {
      const float* src = (const float*)xin + (size_t)(m0 + srow) * EE + k0 + scol;
#pragma unroll
      for (int i = 0; i < 4; ++i) {
        v4f f = *(const v4f*)(src + i * 4);
        v2h p0; p0.x = (_Float16)f.x; p0.y = (_Float16)f.y;
        v2h p1; p1.x = (_Float16)f.z; p1.y = (_Float16)f.w;
        *(v2h*)&As[srow * AP + scol + i * 4]     = p0;
        *(v2h*)&As[srow * AP + scol + i * 4 + 2] = p1;
      }
    }
    // ---- stage W tile (rows = output columns) ----
    {
      const float* src = W + (size_t)(n0 + srow) * EE + k0 + scol;
#pragma unroll
      for (int i = 0; i < 4; ++i) {
        v4f f = *(const v4f*)(src + i * 4);
        v2h p0; p0.x = (_Float16)f.x; p0.y = (_Float16)f.y;
        v2h p1; p1.x = (_Float16)f.z; p1.y = (_Float16)f.w;
        *(v2h*)&Ws[srow * AP + scol + i * 4]     = p0;
        *(v2h*)&Ws[srow * AP + scol + i * 4 + 2] = p1;
      }
    }
    // prefetch next k-slab of A into L2
    if (k0 + KS < EE) {
      const char* pf = (const char*)xin +
          ((size_t)(m0 + srow) * EE + k0 + KS + scol) * (IN_F16 ? 2 : 4);
      __builtin_prefetch(pf, 0, 1);
    }
    __syncthreads();

    v16h af[2];
#pragma unroll
    for (int mi = 0; mi < 2; ++mi)
      af[mi] = frag_ld(&As[(wm + mi * 16) * AP], AP);
#pragma unroll
    for (int ni = 0; ni < 4; ++ni) {
      v16h bf = frag_ld(&Ws[(wn + ni * 16) * AP], AP);
#pragma unroll
      for (int mi = 0; mi < 2; ++mi) acc[mi][ni] = wmma_f16(af[mi], bf, acc[mi][ni]);
    }
    __syncthreads();
  }

  // ---- epilogue: bias + store ----
#pragma unroll
  for (int mi = 0; mi < 2; ++mi) {
#pragma unroll
    for (int ni = 0; ni < 4; ++ni) {
      const int rbase = m0 + wm + mi * 16 + sel * 8;     // global row of acc[..][0]
      const int col = n0 + wn + ni * 16 + lr;            // global col
      const float bv = bias[col];
      if constexpr (OUT_F32) {
        float* o = (float*)out;
#pragma unroll
        for (int r = 0; r < 8; ++r)
          o[(size_t)(rbase + r) * EE + col] = acc[mi][ni][r] + bv;
      } else if constexpr (!TRANS_OUT) {
        _Float16* o = (_Float16*)out;
#pragma unroll
        for (int r = 0; r < 8; ++r)
          o[(size_t)(rbase + r) * EE + col] = (_Float16)(acc[mi][ni][r] + bv);
      } else {
        // transposed per-head V: vT[b][col][token], token runs are contiguous
        _Float16* o = (_Float16*)out;
#pragma unroll
        for (int r = 0; r < 8; r += 2) {
          const int row = rbase + r;
          const int bidx = row / NN;
          const int tok = row - bidx * NN;
          const size_t off = (size_t)bidx * EE * NN + (size_t)col * NN + tok;
          v2h p;
          p.x = (_Float16)(acc[mi][ni][r] + bv);
          p.y = (_Float16)(acc[mi][ni][r + 1] + bv);
          *(v2h*)(o + off) = p;
        }
      }
    }
  }
}

// =====================================================================
// Flash-style attention.  Grid (N/64, H, B), 128 threads = 4 waves,
// each wave owns 16 query rows.  K/vT 64x64 f16 tiles double-buffered in
// LDS via the Tensor Data Mover (LDS row pitch 72 halves via D# padding).
// softmax(S)/sqrt(E) folded into the final O normalization.
// =====================================================================
#define KT  64
#define LP  72   // padded LDS pitch (halves): 128B row + 16B pad

__global__ __launch_bounds__(128) void attention_kernel(
    const _Float16* __restrict__ q16,   // [MM, EE]
    const _Float16* __restrict__ k16,   // [MM, EE]
    const _Float16* __restrict__ vT,    // [b][h*D+d][token]
    _Float16* __restrict__ o16) {       // [MM, EE]
  __shared__ __attribute__((aligned(16))) _Float16 Ks[2][KT * LP];
  __shared__ __attribute__((aligned(16))) _Float16 Vs[2][KT * LP];
  __shared__ __attribute__((aligned(16))) _Float16 Ps[4][16 * LP];

  const int qb = blockIdx.x;       // query tile (64 rows)
  const int h  = blockIdx.y;
  const int b  = blockIdx.z;
  const int t = threadIdx.x, lane = t & 31, w = t >> 5;
  const int sel = lane >> 4, lr = lane & 15;

  const int qrow = qb * 64 + w * 16;   // first query row of this wave (in seq)

  // Q A-fragments straight from global (rows contiguous over d)
  v16h qa[2];
#pragma unroll
  for (int kk = 0; kk < 2; ++kk) {
    const _Float16* p = q16 + (size_t)(b * NN + qrow + lr) * EE + h * DD + kk * 32 + sel * 8;
    union { v16h v; v8h hh[2]; } u;
    u.hh[0] = *(const v8h*)(p);
    u.hh[1] = *(const v8h*)(p + 16);
    qa[kk] = u.v;
  }

  v8f o[4];
#pragma unroll
  for (int oi = 0; oi < 4; ++oi) o[oi] = (v8f){0, 0, 0, 0, 0, 0, 0, 0};
  float mrow[8], lrow[8];
#pragma unroll
  for (int r = 0; r < 8; ++r) { mrow[r] = -1e30f; lrow[r] = 0.0f; }

  const int nkt = NN / KT;   // 32

  // issue tile j into buffer buf (wave0 -> K, wave1 -> vT)
  auto issue = [&](int j, int buf) {
    if (w == 0) {
      const _Float16* g = k16 + (size_t)(b * NN + j * KT) * EE + h * DD;
      tdm_load_2d_f16(lds_offset(&Ks[buf][0]), g, DD, KT, DD, NN, EE, 4, 3, 1);
    } else if (w == 1) {
      const _Float16* g = vT + (size_t)b * EE * NN + (size_t)(h * DD) * NN + j * KT;
      tdm_load_2d_f16(lds_offset(&Vs[buf][0]), g, KT, DD, NN, DD, NN, 4, 3, 1);
    }
  };

  issue(0, 0);

  for (int j = 0; j < nkt; ++j) {
    const int cur = j & 1;
    if (j + 1 < nkt) {
      issue(j + 1, cur ^ 1);
      if (w < 2) __builtin_amdgcn_s_wait_tensorcnt(1);
    } else {
      if (w < 2) __builtin_amdgcn_s_wait_tensorcnt(0);
    }
    __syncthreads();

    const _Float16* Kb = &Ks[cur][0];
    const _Float16* Vb = &Vs[cur][0];

    // ---- S = Q * K^T (16 x 64 strip per wave) ----
    v8f s[4];
#pragma unroll
    for (int ni = 0; ni < 4; ++ni) {
      v16h b0 = frag_ld(Kb + (ni * 16) * LP, LP);        // d = 0..31
      v16h b1 = frag_ld(Kb + (ni * 16) * LP + 32, LP);   // d = 32..63
      v8f c0 = (v8f){0, 0, 0, 0, 0, 0, 0, 0};
      c0 = wmma_f16(qa[0], b0, c0);
      s[ni] = wmma_f16(qa[1], b1, c0);
    }

    // ---- online softmax update ----
    float rmax[8];
#pragma unroll
    for (int r = 0; r < 8; ++r) {
      float v = s[0][r];
#pragma unroll
      for (int ni = 1; ni < 4; ++ni) v = fmaxf(v, s[ni][r]);
#pragma unroll
      for (int msk = 1; msk < 16; msk <<= 1)
        v = fmaxf(v, __shfl_xor(v, msk, 32));
      rmax[r] = v;
    }
    float scale[8];
#pragma unroll
    for (int r = 0; r < 8; ++r) {
      const float mnew = fmaxf(mrow[r], rmax[r]);
      scale[r] = __expf(mrow[r] - mnew);
      mrow[r] = mnew;
    }
    float rsum[8];
#pragma unroll
    for (int r = 0; r < 8; ++r) {
      float acc = 0.0f;
#pragma unroll
      for (int ni = 0; ni < 4; ++ni) {
        const float e = __expf(s[ni][r] - mrow[r]);
        s[ni][r] = e;
        acc += e;
      }
#pragma unroll
      for (int msk = 1; msk < 16; msk <<= 1)
        acc += __shfl_xor(acc, msk, 32);
      rsum[r] = acc;
    }
#pragma unroll
    for (int r = 0; r < 8; ++r) lrow[r] = lrow[r] * scale[r] + rsum[r];
#pragma unroll
    for (int oi = 0; oi < 4; ++oi)
#pragma unroll
      for (int r = 0; r < 8; ++r) o[oi][r] *= scale[r];

    // ---- relayout P (C-frag -> A-frag) via wave-private LDS ----
    _Float16* ps = &Ps[w][0];
#pragma unroll
    for (int ni = 0; ni < 4; ++ni)
#pragma unroll
      for (int r = 0; r < 8; ++r)
        ps[(size_t)(r + sel * 8) * LP + ni * 16 + lr] = (_Float16)s[ni][r];
    asm volatile("s_wait_dscnt 0" ::: "memory");
    v16h pa0 = frag_ld(ps, LP);         // key = 0..31
    v16h pa1 = frag_ld(ps + 32, LP);    // key = 32..63

    // ---- O += P * V ----
#pragma unroll
    for (int oi = 0; oi < 4; ++oi) {
      v16h vb0 = frag_ld(Vb + (oi * 16) * LP, LP);        // key = 0..31
      v16h vb1 = frag_ld(Vb + (oi * 16) * LP + 32, LP);   // key = 32..63
      o[oi] = wmma_f16(pa0, vb0, o[oi]);
      o[oi] = wmma_f16(pa1, vb1, o[oi]);
    }
    __syncthreads();
  }

  // ---- normalize: softmax sum and the post-softmax 1/sqrt(E) quirk ----
  float rinv[8];
#pragma unroll
  for (int r = 0; r < 8; ++r) rinv[r] = 1.0f / (lrow[r] * 32.0f);  // sqrt(1024)=32
#pragma unroll
  for (int oi = 0; oi < 4; ++oi) {
    const int dcol = h * DD + oi * 16 + lr;
#pragma unroll
    for (int r = 0; r < 8; ++r) {
      const int tok = qrow + r + sel * 8;
      o16[(size_t)(b * NN + tok) * EE + dcol] = (_Float16)(o[oi][r] * rinv[r]);
    }
  }
}

// =====================================================================
extern "C" void kernel_launch(void* const* d_in, const int* in_sizes, int n_in,
                              void* d_out, int out_size, void* d_ws, size_t ws_size,
                              hipStream_t stream) {
  const float* queries = (const float*)d_in[0];
  const float* keys    = (const float*)d_in[1];
  const float* values  = (const float*)d_in[2];
  const float* Wq = (const float*)d_in[3];
  const float* bq = (const float*)d_in[4];
  const float* Wk = (const float*)d_in[5];
  const float* bk = (const float*)d_in[6];
  const float* Wv = (const float*)d_in[7];
  const float* bv = (const float*)d_in[8];
  const float* Wp = (const float*)d_in[9];
  const float* bp = (const float*)d_in[10];

  // workspace layout (f16): q | k | vT | attn_out  (4 * 8.4MB = 33.6MB)
  _Float16* q16 = (_Float16*)d_ws;
  _Float16* k16 = q16 + (size_t)MM * EE;
  _Float16* vTw = k16 + (size_t)MM * EE;
  _Float16* a16 = vTw + (size_t)MM * EE;

  dim3 gg(MM / 128, EE / 128);   // 32 x 8
  proj_gemm_kernel<false, false, false><<<gg, 256, 0, stream>>>(queries, Wq, bq, q16);
  proj_gemm_kernel<false, false, false><<<gg, 256, 0, stream>>>(keys,    Wk, bk, k16);
  proj_gemm_kernel<false, true,  false><<<gg, 256, 0, stream>>>(values,  Wv, bv, vTw);

  attention_kernel<<<dim3(NN / 64, HH, BB), 128, 0, stream>>>(q16, k16, vTw, a16);

  proj_gemm_kernel<true, false, true><<<gg, 256, 0, stream>>>(a16, Wp, bp, (float*)d_out);
}
